// MambaMIL_82935818486430
// MI455X (gfx1250) — matrix operand order repaired and checked by reading, not compiled, not run on hardware
//
#include <hip/hip_runtime.h>
#include <hip/hip_bf16.h>

// ---------------------------------------------------------------------------
// MambaMIL forward for MI455X (gfx1250, wave32).
//  - GEMMs: v_wmma_f32_16x16x32_f16, double-buffered LDS tiles fed by the
//    Tensor Data Mover (tensor_load_to_lds + s_wait_tensorcnt), with TDM
//    padding reproducing the bank-conflict-free [rows][KT+2] f16 layout.
//  - All GEMM operands kept as f16 mirrors (f32 masters for scan/LN users).
//  - Selective scan: one lane per (b,d,n) state, 16-lane shfl reduction.
// ---------------------------------------------------------------------------

#define G_BAGS  2
#define L_SEQ   2048
#define IN_SZ   1024
#define DM      512      // d_model
#define DI      1024     // d_inner
#define DT_RANK 32
#define D_STATE 16
#define KCONV   4
#define DEPTH   2
#define XDBL_W  (DT_RANK + 2 * D_STATE)   // 64

typedef __attribute__((ext_vector_type(16))) _Float16 v16h;
typedef __attribute__((ext_vector_type(8)))  float    v8f;
typedef __attribute__((ext_vector_type(4)))  unsigned u32x4;
typedef __attribute__((ext_vector_type(8)))  int      i32x8;
typedef __attribute__((ext_vector_type(4)))  int      i32x4;

#if defined(__HIP_DEVICE_COMPILE__) && __has_builtin(__builtin_amdgcn_tensor_load_to_lds)
#define USE_TDM 1
#else
#define USE_TDM 0
#endif

__device__ __forceinline__ float siluf(float x) { return x / (1.f + __expf(-x)); }

#if USE_TDM
// 2D tile DMA: global (f16, row-major, rowStride halves) -> LDS, with 1-DWORD
// padding after every 16 DWORDs (=> 34-half LDS row pitch for a 32-half tile
// row).  remK/remRows = remaining tensor extent from the tile origin: hardware
// zero-fills reads beyond it.
__device__ __forceinline__ void tdm_load_2d(
    const _Float16* gsrc, unsigned lds_byte_off,
    unsigned tileK, unsigned tileRows,
    unsigned remK, unsigned remRows, unsigned rowStride)
{
  unsigned long long ga = (unsigned long long)(__UINTPTR_TYPE__)gsrc;
  u32x4 g0;
  g0[0] = 1u;                                        // count=1 (valid), user mode
  g0[1] = lds_byte_off;                              // lds_addr [63:32]
  g0[2] = (unsigned)(ga & 0xffffffffu);              // global_addr lo
  g0[3] = (unsigned)((ga >> 32) & 0x01ffffffu) | (2u << 30);   // addr hi | type=2
  i32x8 g1;
  g1[0] = (int)((1u << 16)                           // data_size = 2 bytes
              | (1u << 20)                           // pad_enable
              | (3u << 22)                           // pad_interval = 16 DWORDs
              | (0u << 25));                         // pad_amount = 1 DWORD
  g1[1] = (int)((remK & 0xffffu) << 16);             // tensor_dim0[15:0]
  g1[2] = (int)(((remK >> 16) & 0xffffu) | ((remRows & 0xffffu) << 16));
  g1[3] = (int)(((remRows >> 16) & 0xffffu) | (tileK << 16));   // tile_dim0
  g1[4] = (int)(tileRows & 0xffffu);                 // tile_dim1 (tile_dim2 = 0 -> 2D)
  g1[5] = (int)rowStride;                            // tensor_dim0_stride lo
  g1[6] = 0;                                         // stride0 hi | stride1 lo
  g1[7] = 0;
  i32x4 z4 = {0, 0, 0, 0};
#if defined(__clang_major__) && (__clang_major__ >= 23)
  i32x8 z8 = {0, 0, 0, 0, 0, 0, 0, 0};
  __builtin_amdgcn_tensor_load_to_lds(g0, g1, z4, z4, z8, 0);   // 6-arg toolchain
#else
  __builtin_amdgcn_tensor_load_to_lds(g0, g1, z4, z4, 0);       // 5-arg toolchain
#endif
}
#endif

// ---------------------------------------------------------------------------
// C[M,N] = act(A[M,(lda)] * W[N,K]^T + bias[N]);  A,W are f16; C is f32 with
// optional f16 mirror C16.  ACT: 0=none, 1=relu, 2=softplus.
// ---------------------------------------------------------------------------
template<int ACT>
__global__ __launch_bounds__(256) void gemm_wmma_kernel(
    const _Float16* __restrict__ A, int lda,
    const _Float16* __restrict__ W,
    const float* __restrict__ bias,
    float* __restrict__ C, _Float16* __restrict__ C16,
    int M, int N, int Kd)
{
  constexpr int BM = 128, BN = 64, KT = 32, LDT = KT + 2;
  __shared__ _Float16 smem[2][(BM + BN) * LDT];      // single LDS object (offset 0)

  const int tid  = threadIdx.x;
  const int wv   = tid >> 5;
  const int lane = tid & 31;
  const int m0   = blockIdx.x * BM;
  const int n0   = blockIdx.y * BN;

  v8f acc[BN / 16];
#pragma unroll
  for (int j = 0; j < BN / 16; ++j)
#pragma unroll
    for (int v = 0; v < 8; ++v) acc[j][v] = 0.f;

  const int nk = Kd / KT;

  // ----- prologue: load panel 0 into buffer 0 -----
  {
#if USE_TDM
    if (wv == 0) {
      tdm_load_2d(A + (size_t)m0 * lda, 0u, KT, BM,
                  (unsigned)Kd, (unsigned)(M - m0), (unsigned)lda);
      tdm_load_2d(W + (size_t)n0 * Kd, (unsigned)(BM * LDT * 2), KT, BN,
                  (unsigned)Kd, (unsigned)(N - n0), (unsigned)Kd);
      __builtin_amdgcn_s_wait_tensorcnt(0);
    }
#else
    _Float16* as = &smem[0][0];
    _Float16* bs = &smem[0][BM * LDT];
    for (int i = tid; i < BM * KT; i += 256) {
      int r = i >> 5, kk = i & 31;
      int gr = m0 + r;
      as[r * LDT + kk] = (gr < M) ? A[(size_t)gr * lda + kk] : (_Float16)0.0f;
    }
    for (int i = tid; i < BN * KT; i += 256) {
      int r = i >> 5, kk = i & 31;
      int gn = n0 + r;
      bs[r * LDT + kk] = (gn < N) ? W[(size_t)gn * Kd + kk] : (_Float16)0.0f;
    }
#endif
  }
  __syncthreads();

  for (int it = 0; it < nk; ++it) {
    // ----- issue DMA for the next panel while computing this one -----
    if (it + 1 < nk) {
      int k0 = (it + 1) * KT;
      unsigned base = (unsigned)(((it + 1) & 1) * (BM + BN) * LDT * 2);
#if USE_TDM
      if (wv == 0) {
        tdm_load_2d(A + (size_t)m0 * lda + k0, base, KT, BM,
                    (unsigned)(Kd - k0), (unsigned)(M - m0), (unsigned)lda);
        tdm_load_2d(W + (size_t)n0 * Kd + k0, base + (unsigned)(BM * LDT * 2), KT, BN,
                    (unsigned)(Kd - k0), (unsigned)(N - n0), (unsigned)Kd);
      }
#else
      _Float16* as = &smem[0][0] + base / 2;
      _Float16* bs = as + BM * LDT;
      for (int i = tid; i < BM * KT; i += 256) {
        int r = i >> 5, kk = i & 31;
        int gr = m0 + r;
        as[r * LDT + kk] = (gr < M) ? A[(size_t)gr * lda + k0 + kk] : (_Float16)0.0f;
      }
      for (int i = tid; i < BN * KT; i += 256) {
        int r = i >> 5, kk = i & 31;
        int gn = n0 + r;
        bs[r * LDT + kk] = (gn < N) ? W[(size_t)gn * Kd + k0 + kk] : (_Float16)0.0f;
      }
#endif
    }

    // ----- WMMA on current buffer -----
    {
      const _Float16* as = &smem[it & 1][0];
      const _Float16* bs = as + BM * LDT;
      // A fragment: 16-bit 16x32 layout (lane m = lane%16; K groups split by lane half)
      const int mrow = wv * 16 + (lane & 15);
      const int kb   = (lane >> 4) * 8;
      v16h a;
#pragma unroll
      for (int v = 0; v < 8; ++v) {
        int kk = ((v & 3) << 1) + kb + ((v >> 2) << 4);
        a[2 * v]     = as[mrow * LDT + kk];
        a[2 * v + 1] = as[mrow * LDT + kk + 1];
      }
      const int kb2 = (lane >> 4) * 8;
#pragma unroll
      for (int j = 0; j < BN / 16; ++j) {
        v16h bf;
#pragma unroll
        for (int t = 0; t < 16; ++t) bf[t] = bs[(j * 16 + (lane & 15)) * LDT + kb2 + (t & 7) + ((t >> 3) << 4)];
        acc[j] = __builtin_amdgcn_wmma_f32_16x16x32_f16(
            false, a, false, bf, (short)0, acc[j], false, false);
      }
    }

    if (it + 1 < nk) {
#if USE_TDM
      if (wv == 0) __builtin_amdgcn_s_wait_tensorcnt(0);
#endif
      __syncthreads();
    }
  }

  // ----- epilogue: C/D layout row = v + 8*(lane>=16), col = lane%16 -----
  const int hi = lane >> 4, col = lane & 15;
#pragma unroll
  for (int j = 0; j < BN / 16; ++j) {
    int gn = n0 + j * 16 + col;
    if (gn >= N) continue;
    float bv = bias ? bias[gn] : 0.f;
#pragma unroll
    for (int v = 0; v < 8; ++v) {
      int gr = m0 + wv * 16 + hi * 8 + v;
      if (gr < M) {
        float x = acc[j][v] + bv;
        if (ACT == 1) x = fmaxf(x, 0.f);
        if (ACT == 2) x = (x > 20.f) ? x : log1pf(__expf(x));
        size_t idx = (size_t)gr * N + gn;
        C[idx] = x;
        if (C16) C16[idx] = (_Float16)x;
      }
    }
  }
}

// ---------------------------------------------------------------------------
// res += x (if res); LN(res or x)*w+b -> out (f32, optional) and out16 (f16,
// optional).  Width fixed at 512; one wave per row, 8 rows / block.
// ---------------------------------------------------------------------------
__global__ __launch_bounds__(256) void resid_ln_kernel(
    const float* __restrict__ x, float* res,
    const float* __restrict__ w, const float* __restrict__ b,
    float* __restrict__ out, _Float16* __restrict__ out16, int rows)
{
  const int row  = blockIdx.x * 8 + (threadIdx.x >> 5);
  const int lane = threadIdx.x & 31;
  if (row >= rows) return;
  float v[16];
  float s = 0.f;
#pragma unroll
  for (int i = 0; i < 16; ++i) {
    int c = lane + 32 * i;
    size_t idx = (size_t)row * DM + c;
    float t = x[idx];
    if (res) { t += res[idx]; res[idx] = t; }
    v[i] = t; s += t;
  }
#pragma unroll
  for (int m = 16; m >= 1; m >>= 1) s += __shfl_xor(s, m, 32);
  float mu = s * (1.f / DM);
  float s2 = 0.f;
#pragma unroll
  for (int i = 0; i < 16; ++i) { float d = v[i] - mu; s2 += d * d; }
#pragma unroll
  for (int m = 16; m >= 1; m >>= 1) s2 += __shfl_xor(s2, m, 32);
  float rstd = rsqrtf(s2 * (1.f / DM) + 1e-5f);
#pragma unroll
  for (int i = 0; i < 16; ++i) {
    int c = lane + 32 * i;
    float o = (v[i] - mu) * rstd * w[c] + b[c];
    size_t idx = (size_t)row * DM + c;
    if (out)   out[idx]   = o;
    if (out16) out16[idx] = (_Float16)o;
  }
}

// ---------------------------------------------------------------------------
// Depthwise causal conv (K=4) + SiLU over the x half of xz; dir=1 runs on the
// L-flipped sequence via index remap (no materialized flip).  Writes f32 (for
// the scan) and f16 (for the x_proj WMMA GEMM).
// ---------------------------------------------------------------------------
__global__ __launch_bounds__(256) void conv_silu_kernel(
    const float* __restrict__ xz, const float* __restrict__ cw,
    const float* __restrict__ cb, float* __restrict__ xc,
    _Float16* __restrict__ xc16, int Bn, int Ls, int dir)
{
  int tid = blockIdx.x * 256 + threadIdx.x;
  int total = Bn * Ls * DI;
  if (tid >= total) return;
  int d = tid & (DI - 1);
  int l = (tid >> 10) % Ls;
  int b = tid / (Ls * DI);
  float acc = cb[d];
#pragma unroll
  for (int k = 0; k < KCONV; ++k) {
    int j = l - (KCONV - 1) + k;
    if (j >= 0) {
      int src = dir ? (Ls - 1 - j) : j;
      acc += cw[d * KCONV + k] * xz[((size_t)b * Ls + src) * (2 * DI) + d];
    }
  }
  float o = siluf(acc);
  xc[tid] = o;
  xc16[tid] = (_Float16)o;
}

// ---------------------------------------------------------------------------
// Selective scan.  Lane owns one (b,d,n) state; sum_n h*C via 16-lane
// shfl_xor; lane n==0 writes y=(sum + D*u)*silu(z) at the un-flipped
// position.  dir=0 stores; dir=1 accumulates and emits the f16 mirror that
// feeds the out_proj WMMA GEMM.
// ---------------------------------------------------------------------------
__global__ __launch_bounds__(256) void scan_kernel(
    const float* __restrict__ u, const float* __restrict__ dlt,
    const float* __restrict__ xdbl, const float* __restrict__ xz,
    const float* __restrict__ A_log, const float* __restrict__ Dv,
    float* __restrict__ y, _Float16* __restrict__ y16,
    int Bn, int Ls, int dir)
{
  int tid = blockIdx.x * 256 + threadIdx.x;
  if (tid >= Bn * DI * D_STATE) return;
  int n = tid & 15;
  int d = (tid >> 4) & (DI - 1);
  int b = tid >> 14;                 // DI*D_STATE == 1<<14 lanes per batch
  const float Acoef = -__expf(A_log[d * D_STATE + n]);
  const float Dd    = Dv[d];
  float h = 0.f;
  for (int l = 0; l < Ls; ++l) {
    size_t rowd = ((size_t)b * Ls + l) * DI + d;
    size_t rowr = ((size_t)b * Ls + l) * XDBL_W;
    float de = dlt[rowd];
    float uu = u[rowd];
    float Bv = xdbl[rowr + DT_RANK + n];
    float Cv = xdbl[rowr + DT_RANK + D_STATE + n];
    h = __expf(de * Acoef) * h + de * uu * Bv;
    float p = h * Cv;
    p += __shfl_xor(p, 1, 32);
    p += __shfl_xor(p, 2, 32);
    p += __shfl_xor(p, 4, 32);
    p += __shfl_xor(p, 8, 32);
    if (n == 0) {
      int lo = dir ? (Ls - 1 - l) : l;
      float zz = xz[((size_t)b * Ls + lo) * (2 * DI) + DI + d];
      float val = (p + Dd * uu) * siluf(zz);
      size_t oi = ((size_t)b * Ls + lo) * DI + d;
      if (dir) {
        float tot = y[oi] + val;
        y[oi] = tot;
        y16[oi] = (_Float16)tot;
      } else {
        y[oi] = val;
      }
    }
  }
}

__global__ __launch_bounds__(256) void mean_kernel(
    const float* __restrict__ po, float* __restrict__ out, int Gn, int Ln, int Cn)
{
  int idx = blockIdx.x * 256 + threadIdx.x;
  if (idx >= Gn * Cn) return;
  int g = idx / Cn, c = idx % Cn;
  float s = 0.f;
  for (int l = 0; l < Ln; ++l) s += po[((size_t)g * Ln + l) * Cn + c];
  out[idx] = s / (float)Ln;
}

__global__ __launch_bounds__(256) void cvt_f16_kernel(
    const float* __restrict__ src, _Float16* __restrict__ dst, int n)
{
  int i = blockIdx.x * 256 + threadIdx.x;
  if (i < n) dst[i] = (_Float16)src[i];
}

// ---------------------------------------------------------------------------
extern "C" void kernel_launch(void* const* d_in, const int* in_sizes, int n_in,
                              void* d_out, int out_size, void* d_ws, size_t ws_size,
                              hipStream_t stream)
{
  (void)in_sizes; (void)n_in; (void)out_size; (void)ws_size;
  const float* feats   = (const float*)d_in[0];
  const float* fc1_w   = (const float*)d_in[1];
  const float* fc1_b   = (const float*)d_in[2];
  const float* in_w    = (const float*)d_in[3];
  const float* cw      = (const float*)d_in[4];
  const float* cb      = (const float*)d_in[5];
  const float* xpw     = (const float*)d_in[6];
  const float* dtw     = (const float*)d_in[7];
  const float* dtb     = (const float*)d_in[8];
  const float* Alog    = (const float*)d_in[9];
  const float* Dvp     = (const float*)d_in[10];
  const float* cwb     = (const float*)d_in[11];
  const float* cbb     = (const float*)d_in[12];
  const float* xpwb    = (const float*)d_in[13];
  const float* dtwb    = (const float*)d_in[14];
  const float* dtbb    = (const float*)d_in[15];
  const float* Alogb   = (const float*)d_in[16];
  const float* Dvpb    = (const float*)d_in[17];
  const float* ow      = (const float*)d_in[18];
  const float* lnw     = (const float*)d_in[19];
  const float* lnb     = (const float*)d_in[20];
  const float* normf_w = (const float*)d_in[21];
  const float* normf_b = (const float*)d_in[22];
  const float* norm_w  = (const float*)d_in[23];
  const float* norm_b  = (const float*)d_in[24];

  float* ws = (float*)d_ws;
  size_t off = 0;
  auto alloc   = [&](size_t n) { float* p = ws + off; off += n; return p; };
  auto alloc16 = [&](size_t n) { return (_Float16*)alloc((n + 1) / 2); };

  const int GL = G_BAGS * L_SEQ;                       // 4096 token rows
  // f32 masters
  float* hbuf = alloc((size_t)GL * DM);
  float* res  = alloc((size_t)GL * DM);
  float* xz   = alloc((size_t)GL * 2 * DI);
  float* xc   = alloc((size_t)GL * DI);
  float* xdb  = alloc((size_t)GL * XDBL_W);
  float* dlt  = alloc((size_t)GL * DI);
  float* ybuf = alloc((size_t)GL * DI);
  // f16 mirrors (GEMM operands)
  _Float16* feats16 = alloc16((size_t)GL * IN_SZ);
  _Float16* fc1w16  = alloc16((size_t)DM * IN_SZ);
  _Float16* inw16   = alloc16((size_t)DEPTH * 2 * DI * DM);
  _Float16* xpw16   = alloc16((size_t)DEPTH * XDBL_W * DI);
  _Float16* dtw16   = alloc16((size_t)DEPTH * DI * DT_RANK);
  _Float16* xpwb16  = alloc16((size_t)DEPTH * XDBL_W * DI);
  _Float16* dtwb16  = alloc16((size_t)DEPTH * DI * DT_RANK);
  _Float16* ow16    = alloc16((size_t)DEPTH * DM * DI);
  _Float16* hn16    = alloc16((size_t)GL * DM);
  _Float16* xc16    = alloc16((size_t)GL * DI);
  _Float16* xdb16   = alloc16((size_t)GL * XDBL_W);
  _Float16* y16     = alloc16((size_t)GL * DI);
  // region-pass (tiny)
  float* rh    = alloc((size_t)G_BAGS * DM);
  float* rres  = alloc((size_t)G_BAGS * DM);
  float* rxz   = alloc((size_t)G_BAGS * 2 * DI);
  float* rxc   = alloc((size_t)G_BAGS * DI);
  float* rxdb  = alloc((size_t)G_BAGS * XDBL_W);
  float* rdlt  = alloc((size_t)G_BAGS * DI);
  float* rybuf = alloc((size_t)G_BAGS * DI);
  float* rtmp  = alloc((size_t)G_BAGS * DM);
  _Float16* rhn16  = alloc16((size_t)G_BAGS * DM);
  _Float16* rxc16  = alloc16((size_t)G_BAGS * DI);
  _Float16* rxdb16 = alloc16((size_t)G_BAGS * XDBL_W);
  _Float16* ry16   = alloc16((size_t)G_BAGS * DI);

  auto cvt = [&](const float* s, _Float16* d, size_t n) {
    cvt_f16_kernel<<<(unsigned)((n + 255) / 256), 256, 0, stream>>>(s, d, (int)n);
  };
  // one-time f32 -> f16 mirrors of all WMMA operand weights + input features
  cvt(feats, feats16, (size_t)GL * IN_SZ);
  cvt(fc1_w, fc1w16, (size_t)DM * IN_SZ);
  cvt(in_w,  inw16,  (size_t)DEPTH * 2 * DI * DM);
  cvt(xpw,   xpw16,  (size_t)DEPTH * XDBL_W * DI);
  cvt(dtw,   dtw16,  (size_t)DEPTH * DI * DT_RANK);
  cvt(xpwb,  xpwb16, (size_t)DEPTH * XDBL_W * DI);
  cvt(dtwb,  dtwb16, (size_t)DEPTH * DI * DT_RANK);
  cvt(ow,    ow16,   (size_t)DEPTH * DM * DI);

  auto gemm = [&](int act, const _Float16* A, int lda, const _Float16* W,
                  const float* bias, float* C, _Float16* C16, int M, int N, int Kd) {
    dim3 grid((M + 127) / 128, N / 64);
    if (act == 0)      gemm_wmma_kernel<0><<<grid, 256, 0, stream>>>(A, lda, W, bias, C, C16, M, N, Kd);
    else if (act == 1) gemm_wmma_kernel<1><<<grid, 256, 0, stream>>>(A, lda, W, bias, C, C16, M, N, Kd);
    else               gemm_wmma_kernel<2><<<grid, 256, 0, stream>>>(A, lda, W, bias, C, C16, M, N, Kd);
  };

  // run_layers: res = h + res; hn = LN(res); h = Mamba(hn); PO = LN_f(h + res)
  auto run_stack = [&](float* H, float* RES, _Float16* HN16, float* XZ,
                       float* XC, _Float16* XC16, float* XDB, _Float16* XDB16,
                       float* DLT, float* Y, _Float16* Y16, float* PO,
                       int Bn, int Ls) {
    int rows = Bn * Ls;
    int lnGrid = (rows + 7) / 8;
    (void)hipMemsetAsync(RES, 0, (size_t)rows * DM * sizeof(float), stream);
    for (int ly = 0; ly < DEPTH; ++ly) {
      const _Float16* in_wl = inw16 + (size_t)ly * 2 * DI * DM;
      const _Float16* owl   = ow16  + (size_t)ly * DM * DI;
      resid_ln_kernel<<<lnGrid, 256, 0, stream>>>(H, RES, lnw + ly * DM, lnb + ly * DM,
                                                  nullptr, HN16, rows);
      gemm(0, HN16, DM, in_wl, nullptr, XZ, nullptr, rows, 2 * DI, DM);
      for (int dir = 0; dir < 2; ++dir) {
        const float*    cwl  = (dir ? cwb   : cw)    + (size_t)ly * DI * KCONV;
        const float*    cbl  = (dir ? cbb   : cb)    + (size_t)ly * DI;
        const _Float16* xpwl = (dir ? xpwb16 : xpw16) + (size_t)ly * XDBL_W * DI;
        const _Float16* dtwl = (dir ? dtwb16 : dtw16) + (size_t)ly * DI * DT_RANK;
        const float*    dtbl = (dir ? dtbb  : dtb)   + (size_t)ly * DI;
        const float*    All  = (dir ? Alogb : Alog)  + (size_t)ly * DI * D_STATE;
        const float*    Dl   = (dir ? Dvpb  : Dvp)   + (size_t)ly * DI;
        int cvTot = rows * DI;
        conv_silu_kernel<<<(cvTot + 255) / 256, 256, 0, stream>>>(XZ, cwl, cbl, XC, XC16, Bn, Ls, dir);
        gemm(0, XC16, DI, xpwl, nullptr, XDB, XDB16, rows, XDBL_W, DI);   // x_proj
        gemm(2, XDB16, XDBL_W, dtwl, dtbl, DLT, nullptr, rows, DI, DT_RANK); // dt+softplus
        int scTot = Bn * DI * D_STATE;
        scan_kernel<<<(scTot + 255) / 256, 256, 0, stream>>>(XC, DLT, XDB, XZ, All, Dl,
                                                             Y, Y16, Bn, Ls, dir);
      }
      gemm(0, Y16, DI, owl, nullptr, H, nullptr, rows, DM, DI);           // out_proj
    }
    resid_ln_kernel<<<lnGrid, 256, 0, stream>>>(H, RES, normf_w, normf_b, PO, nullptr, rows);
  };

  // 1) fc1 + ReLU: h0 = relu(feats @ fc1_w^T + b)
  gemm(1, feats16, IN_SZ, fc1w16, fc1_b, hbuf, nullptr, GL, DM, IN_SZ);

  // 2) patch-level Mamba stack; patch_out written into ybuf (dead by then)
  run_stack(hbuf, res, hn16, xz, xc, xc16, xdb, xdb16, dlt, ybuf, y16, ybuf, G_BAGS, L_SEQ);

  // 3) region means: [G, 512]
  mean_kernel<<<(G_BAGS * DM + 255) / 256, 256, 0, stream>>>(ybuf, rh, G_BAGS, L_SEQ, DM);

  // 4) region-level stack over sequence of length G
  run_stack(rh, rres, rhn16, rxz, rxc, rxc16, rxdb, rxdb16, rdlt, rybuf, ry16, rtmp, 1, G_BAGS);

  // 5) final LayerNorm -> d_out [1, G, 512]
  resid_ln_kernel<<<1, 256, 0, stream>>>(rtmp, nullptr, norm_w, norm_b, (float*)d_out, nullptr, G_BAGS);
}
